// DecoderBlock_71794673320657
// MI455X (gfx1250) — matrix operand order repaired and checked by reading, not compiled, not run on hardware
//
#include <hip/hip_runtime.h>
#include <hip/hip_bf16.h>
#include <stdint.h>

// ---------------------------------------------------------------------------
// CDNA5 (gfx1250) wave32 WMMA types / helpers
// ---------------------------------------------------------------------------
typedef __attribute__((ext_vector_type(16))) __bf16 bf16x16;
typedef __attribute__((ext_vector_type(8)))  float  f32x8;

union BF16Frag { uint32_t u[8]; bf16x16 v; };

__device__ __forceinline__ uint16_t f2bf(float f) {
  uint32_t u = __float_as_uint(f);
  u += 0x7fffu + ((u >> 16) & 1u);   // round-to-nearest-even
  return (uint16_t)(u >> 16);
}

__device__ __forceinline__ uint32_t pack2bf(float a, float b) {
  return (uint32_t)f2bf(a) | ((uint32_t)f2bf(b) << 16);
}

__device__ __forceinline__ f32x8 wmma_bf16(bf16x16 a, bf16x16 b, f32x8 c) {
  // v_wmma_f32_16x16x32_bf16 (8-arg: neg_a, A, neg_b, B, c_mod, C, reuse_a, reuse_b)
  return __builtin_amdgcn_wmma_f32_16x16x32_bf16(false, a, false, b, (short)0, c, false, false);
}

// Low 32 bits of a flat shared-memory address == LDS byte offset (aperture
// base has zero low bits); this is what async-to-LDS VDST expects.
__device__ __forceinline__ uint32_t lds_off(const void* p) {
  return (uint32_t)(uintptr_t)p;
}

// Async DMA: global (16B per lane) -> LDS, tracked by ASYNCcnt.
__device__ __forceinline__ void async_ld_b128(uint32_t lds, const void* gaddr) {
  asm volatile("global_load_async_to_lds_b128 %0, %1, off"
               :: "v"(lds), "v"(gaddr) : "memory");
}
__device__ __forceinline__ void wait_async0() {
  asm volatile("s_wait_asynccnt 0" ::: "memory");
}

// A-fragment (16x32 bf16). rowp = pointer to (this lane's row, k=0).
// ISA layout: lanes 0-15: K {0..7,16..23}; lanes 16-31: K {8..15,24..31}.
__device__ __forceinline__ bf16x16 frag_a(const uint16_t* rowp, int lane) {
  const int kb = (lane & 16) ? 8 : 0;
  BF16Frag f;
#pragma unroll
  for (int j = 0; j < 4; ++j) {
    f.u[j]     = *(const uint32_t*)(rowp + kb + 2 * j);
    f.u[4 + j] = *(const uint32_t*)(rowp + 16 + kb + 2 * j);
  }
  return f.v;
}

// B-fragment (32x16 bf16), source stored [n][k] row-major; rowp = row n.
// ISA layout: lanes 0-15: K {0..7,16..23}; lanes 16-31: K {8..15,24..31}.
__device__ __forceinline__ bf16x16 frag_b(const uint16_t* rowp, int lane) {
  const int kb = (lane & 16) ? 8 : 0;
  BF16Frag f;
#pragma unroll
  for (int j = 0; j < 4; ++j) {
    f.u[j]     = *(const uint32_t*)(rowp + kb + 2 * j);
    f.u[4 + j] = *(const uint32_t*)(rowp + 16 + kb + 2 * j);
  }
  return f.v;
}

// ---------------------------------------------------------------------------
// fp32 -> bf16 conversion, vectorized (n multiple of 4 for all our arrays)
// ---------------------------------------------------------------------------
__global__ void cvt_f32_bf16(const float4* __restrict__ s, uint2* __restrict__ d, int n4) {
  for (int i = blockIdx.x * blockDim.x + threadIdx.x; i < n4; i += gridDim.x * blockDim.x) {
    const float4 x = s[i];
    d[i] = make_uint2(pack2bf(x.x, x.y), pack2bf(x.z, x.w));
  }
}

// ---------------------------------------------------------------------------
// LayerNorm over 1024-wide rows; one wave32 per row, 8 rows per block.
// float4 loads, packed bf16 stores.
// ---------------------------------------------------------------------------
__global__ __launch_bounds__(256) void ln_bf16(const float* __restrict__ X,
                                               const float* __restrict__ G,
                                               const float* __restrict__ Bt,
                                               uint16_t* __restrict__ Y) {
  const int NC = 1024;
  const int lane = threadIdx.x & 31;
  const int row  = blockIdx.x * 8 + (threadIdx.x >> 5);
  const float4* xr = (const float4*)(X + (size_t)row * NC);
  float4 v[8];
  float s = 0.f, sq = 0.f;
#pragma unroll
  for (int i = 0; i < 8; ++i) {
    v[i] = xr[lane + 32 * i];
    s  += v[i].x + v[i].y + v[i].z + v[i].w;
    sq += v[i].x * v[i].x + v[i].y * v[i].y + v[i].z * v[i].z + v[i].w * v[i].w;
  }
#pragma unroll
  for (int d = 1; d < 32; d <<= 1) { s += __shfl_xor(s, d, 32); sq += __shfl_xor(sq, d, 32); }
  const float mu  = s * (1.f / 1024.f);
  const float var = sq * (1.f / 1024.f) - mu * mu;
  const float rs  = rsqrtf(var + 1e-5f);
  const float4* G4 = (const float4*)G;
  const float4* B4 = (const float4*)Bt;
  uint2* yr = (uint2*)(Y + (size_t)row * NC);
#pragma unroll
  for (int i = 0; i < 8; ++i) {
    const int c4 = lane + 32 * i;
    const float4 g = G4[c4], b = B4[c4];
    const float a0 = (v[i].x - mu) * rs * g.x + b.x;
    const float a1 = (v[i].y - mu) * rs * g.y + b.y;
    const float a2 = (v[i].z - mu) * rs * g.z + b.z;
    const float a3 = (v[i].w - mu) * rs * g.w + b.w;
    yr[c4] = make_uint2(pack2bf(a0, a1), pack2bf(a2, a3));
  }
}

// ---------------------------------------------------------------------------
// Generic GEMM: C[M,N] = A[M,K] @ W[N,K]^T (torch Linear), bf16 in, f32 acc.
// Block tile 128x128, BK=32, 8 waves each own 32x64 (2x4 WMMA tiles).
// Double-buffered LDS filled by GLOBAL_LOAD_ASYNC_TO_LDS_B128 (ASYNCcnt),
// one barrier per K-step; tile k+1 DMA overlaps tile-k WMMAs.
// Compile-time epilogue: bias / ReLU / residual / f32-or-bf16 output.
// ---------------------------------------------------------------------------
#define GK 40  // padded LDS k-stride: 32 + 8 (80B rows, 16B aligned)

template <int OUT_BF16, int RELU, int BIAS, int RES>
__global__ __launch_bounds__(256) void gemm_bf16_t(
    const uint16_t* __restrict__ A, int lda,
    const uint16_t* __restrict__ W, int ldw,
    float* __restrict__ Cf, uint16_t* __restrict__ Cb, int ldc,
    const float* __restrict__ bias, const float* __restrict__ res,
    int K) {
  __shared__ uint16_t As[2][128 * GK];
  __shared__ uint16_t Bs[2][128 * GK];
  const int tid = threadIdx.x, lane = tid & 31, wid = tid >> 5;
  const int bm = blockIdx.y * 128, bn = blockIdx.x * 128;
  const int wm = (wid >> 1) * 32, wn = (wid & 1) * 64;

  // Per-thread tile-copy coords: two 16B chunks per matrix per K-step.
  const int r0 = tid >> 2, s0 = (tid & 3) * 8, r1 = r0 + 64;
  const uint16_t* gA0 = A + (size_t)(bm + r0) * lda + s0;
  const uint16_t* gA1 = A + (size_t)(bm + r1) * lda + s0;
  const uint16_t* gW0 = W + (size_t)(bn + r0) * ldw + s0;
  const uint16_t* gW1 = W + (size_t)(bn + r1) * ldw + s0;
  const uint32_t lA0[2] = { lds_off(&As[0][r0 * GK + s0]), lds_off(&As[1][r0 * GK + s0]) };
  const uint32_t lA1[2] = { lds_off(&As[0][r1 * GK + s0]), lds_off(&As[1][r1 * GK + s0]) };
  const uint32_t lW0[2] = { lds_off(&Bs[0][r0 * GK + s0]), lds_off(&Bs[1][r0 * GK + s0]) };
  const uint32_t lW1[2] = { lds_off(&Bs[0][r1 * GK + s0]), lds_off(&Bs[1][r1 * GK + s0]) };

  auto issue = [&](int buf, int k0) {
    async_ld_b128(lA0[buf], gA0 + k0);
    async_ld_b128(lA1[buf], gA1 + k0);
    async_ld_b128(lW0[buf], gW0 + k0);
    async_ld_b128(lW1[buf], gW1 + k0);
  };

  const f32x8 vzero = {0.f, 0.f, 0.f, 0.f, 0.f, 0.f, 0.f, 0.f};
  f32x8 acc[2][4];
#pragma unroll
  for (int mt = 0; mt < 2; ++mt)
#pragma unroll
    for (int nt = 0; nt < 4; ++nt) acc[mt][nt] = vzero;

  issue(0, 0);
  int buf = 0;
  for (int k0 = 0; k0 < K; k0 += 32, buf ^= 1) {
    wait_async0();      // this wave's current-buffer DMA complete
    __syncthreads();    // all waves resident; all prior-buffer reads retired
    if (k0 + 32 < K) issue(buf ^ 1, k0 + 32);   // prefetch next tile

    bf16x16 af[2], bfr[4];
#pragma unroll
    for (int mt = 0; mt < 2; ++mt)
      af[mt]  = frag_a(&As[buf][(wm + mt * 16 + (lane & 15)) * GK], lane);
#pragma unroll
    for (int nt = 0; nt < 4; ++nt)
      bfr[nt] = frag_b(&Bs[buf][(wn + nt * 16 + (lane & 15)) * GK], lane);
#pragma unroll
    for (int mt = 0; mt < 2; ++mt)
#pragma unroll
      for (int nt = 0; nt < 4; ++nt) acc[mt][nt] = wmma_bf16(af[mt], bfr[nt], acc[mt][nt]);
  }

  const int half8 = (lane >> 4) << 3;
  const int nIn   = lane & 15;
#pragma unroll
  for (int mt = 0; mt < 2; ++mt)
#pragma unroll
    for (int nt = 0; nt < 4; ++nt) {
      const int col = bn + wn + nt * 16 + nIn;
      float bv = 0.f;
      if (BIAS) bv = bias[col];
#pragma unroll
      for (int r = 0; r < 8; ++r) {
        const int row = bm + wm + mt * 16 + half8 + r;
        float v = acc[mt][nt][r] + bv;
        if (RELU) v = fmaxf(v, 0.f);
        if (RES)  v += res[(size_t)row * ldc + col];
        if (OUT_BF16) Cb[(size_t)row * ldc + col] = f2bf(v);
        else          Cf[(size_t)row * ldc + col] = v;
      }
    }
}

// ---------------------------------------------------------------------------
// Flash attention (bf16 QKV in [tok, H] layout with head slices, DK=128).
// One workgroup = (b, h, 128 q rows); 8 waves x 16 q rows. Online softmax,
// K tile (64x128) + V tile transposed (128x64) + per-wave P staging in LDS.
// ---------------------------------------------------------------------------
#define KSS 136  // K tile dk-stride (128 + 8)
#define VTS 72   // V^T tile key-stride (64 + 8)
#define PSS 72   // P staging key-stride (64 + 8)

template <int CAUSAL>
__global__ __launch_bounds__(256) void flash_attn(
    const uint16_t* __restrict__ Q, const uint16_t* __restrict__ Kp,
    const uint16_t* __restrict__ Vp, uint16_t* __restrict__ O,
    int Tq, int Tk) {
  __shared__ uint16_t Ks[64 * KSS];
  __shared__ uint16_t Vt[128 * VTS];
  __shared__ uint16_t Ps[8 * 16 * PSS];
  const int NHh = 8, DKH = 128, Hn = 1024;
  const int tid = threadIdx.x, lane = tid & 31, wid = tid >> 5;
  const int b = blockIdx.y / NHh, h = blockIdx.y % NHh;
  const int q0 = blockIdx.x * 128;
  const size_t qbase  = ((size_t)b * Tq + q0) * Hn + h * DKH;
  const size_t kvbase = ((size_t)b * Tk) * Hn + h * DKH;

  bf16x16 aq[4];                       // this wave's 16 q rows, dk split 4x32
  {
    const uint16_t* qp = Q + qbase + (size_t)(wid * 16 + (lane & 15)) * Hn;
#pragma unroll
    for (int kc = 0; kc < 4; ++kc) aq[kc] = frag_a(qp + kc * 32, lane);
  }

  const f32x8 vzero = {0.f, 0.f, 0.f, 0.f, 0.f, 0.f, 0.f, 0.f};
  f32x8 oacc[8];
  float m_r[8], l_r[8];
#pragma unroll
  for (int d = 0; d < 8; ++d) oacc[d] = vzero;
#pragma unroll
  for (int r = 0; r < 8; ++r) { m_r[r] = -3.0e38f; l_r[r] = 0.f; }

  const float scale = 0.08838834764831845f;   // 1/sqrt(128)
  const float L2E   = 1.4426950408889634f;
  int ktEnd = CAUSAL ? (q0 + 128) : Tk;
  if (ktEnd > Tk) ktEnd = Tk;

  for (int kt = 0; kt < ktEnd; kt += 64) {
    __syncthreads();
#pragma unroll
    for (int i = 0; i < 4; ++i) {            // K tile: 64x128 bf16, row-major
      const int idx = tid + i * 256;
      const int row = idx >> 4, seg = (idx & 15) * 8;
      *(uint4*)(Ks + row * KSS + seg) =
          *(const uint4*)(Kp + kvbase + (size_t)(kt + row) * Hn + seg);
    }
#pragma unroll
    for (int i = 0; i < 16; ++i) {           // V tile transposed -> Vt[dk][key]
      const int idx = tid + i * 256;
      const int key = idx >> 6, dk2 = (idx & 63) * 2;
      const uint32_t dv = *(const uint32_t*)(Vp + kvbase + (size_t)(kt + key) * Hn + dk2);
      Vt[dk2 * VTS + key]       = (uint16_t)(dv & 0xffffu);
      Vt[(dk2 + 1) * VTS + key] = (uint16_t)(dv >> 16);
    }
    __syncthreads();

    // S = Q @ K^T  (16 q rows x 64 keys per wave)
    f32x8 s[4];
#pragma unroll
    for (int nt = 0; nt < 4; ++nt) s[nt] = vzero;
#pragma unroll
    for (int nt = 0; nt < 4; ++nt) {
      const uint16_t* kr = Ks + (nt * 16 + (lane & 15)) * KSS;
#pragma unroll
      for (int kc = 0; kc < 4; ++kc)
        s[nt] = wmma_bf16(aq[kc], frag_b(kr + kc * 32, lane), s[nt]);
    }

    // scale + causal mask + online max update (row = vgpr r + 8*(lane>=16))
    const int colB = kt + (lane & 15);
    const int rowB = q0 + wid * 16 + ((lane >> 4) << 3);
    float alpha[8];
#pragma unroll
    for (int r = 0; r < 8; ++r) {
      float mx = -3.0e38f;
#pragma unroll
      for (int nt = 0; nt < 4; ++nt) {
        float v = s[nt][r] * scale;
        if (CAUSAL && (colB + nt * 16) > (rowB + r)) v = -3.0e38f;
        s[nt][r] = v;
        mx = fmaxf(mx, v);
      }
#pragma unroll
      for (int d = 1; d < 16; d <<= 1) mx = fmaxf(mx, __shfl_xor(mx, d, 32));
      const float mn = fmaxf(m_r[r], mx);
      alpha[r] = exp2f((m_r[r] - mn) * L2E);
      m_r[r]   = mn;
    }

    // P = exp(S - m); stage to LDS (C-layout -> A-layout), rescale l and O
    uint16_t* pw = Ps + wid * 16 * PSS;
#pragma unroll
    for (int r = 0; r < 8; ++r) {
      float rs = 0.f;
      const int prow = ((lane >> 4) << 3) + r;
#pragma unroll
      for (int nt = 0; nt < 4; ++nt) {
        const float p = exp2f((s[nt][r] - m_r[r]) * L2E);
        rs += p;
        pw[prow * PSS + nt * 16 + (lane & 15)] = f2bf(p);
      }
#pragma unroll
      for (int d = 1; d < 16; d <<= 1) rs += __shfl_xor(rs, d, 32);
      l_r[r] = l_r[r] * alpha[r] + rs;
#pragma unroll
      for (int dt = 0; dt < 8; ++dt) oacc[dt][r] *= alpha[r];
    }

    // O += P @ V   (key chunks of 32, dk tiles of 16)
#pragma unroll
    for (int kc2 = 0; kc2 < 2; ++kc2) {
      const bf16x16 pa = frag_a(pw + (lane & 15) * PSS + kc2 * 32, lane);
#pragma unroll
      for (int dt = 0; dt < 8; ++dt) {
        const uint16_t* vr = Vt + (dt * 16 + (lane & 15)) * VTS + kc2 * 32;
        oacc[dt] = wmma_bf16(pa, frag_b(vr, lane), oacc[dt]);
      }
    }
  }

  // normalize and write context bf16 at [tok, h*128 + dk]
  const size_t obase = ((size_t)b * Tq + q0 + wid * 16) * Hn + h * DKH;
#pragma unroll
  for (int r = 0; r < 8; ++r) {
    const float inv = 1.f / l_r[r];
    const int rw = ((lane >> 4) << 3) + r;
#pragma unroll
    for (int dt = 0; dt < 8; ++dt)
      O[obase + (size_t)rw * Hn + dt * 16 + (lane & 15)] = f2bf(oacc[dt][r] * inv);
  }
}

// ---------------------------------------------------------------------------
// Host orchestration
// ---------------------------------------------------------------------------
extern "C" void kernel_launch(void* const* d_in, const int* in_sizes, int n_in,
                              void* d_out, int out_size, void* d_ws, size_t ws_size,
                              hipStream_t stream) {
  (void)in_sizes; (void)n_in; (void)out_size; (void)ws_size;
  const int T = 1024, TE = 1024, H = 1024, NH = 8, F = 4096, Bn = 8;
  const int M  = Bn * T;   // 8192 decoder token rows
  const int ME = Bn * TE;  // 8192 encoder token rows

  const float* input_ = (const float*)d_in[0];
  const float* enc    = (const float*)d_in[1];
  // d_in[2], d_in[3]: masks (causal triu / all-false) -> computed analytically
  const float* Wq_s = (const float*)d_in[4];
  const float* Wk_s = (const float*)d_in[5];
  const float* Wv_s = (const float*)d_in[6];
  const float* Wo_s = (const float*)d_in[7];
  const float* Wq_c = (const float*)d_in[8];
  const float* Wk_c = (const float*)d_in[9];
  const float* Wv_c = (const float*)d_in[10];
  const float* Wo_c = (const float*)d_in[11];
  const float* w1 = (const float*)d_in[12];
  const float* b1 = (const float*)d_in[13];
  const float* w2 = (const float*)d_in[14];
  const float* b2 = (const float*)d_in[15];
  const float* g_mmha = (const float*)d_in[16];
  const float* b_mmha = (const float*)d_in[17];
  const float* g_mha  = (const float*)d_in[18];
  const float* b_mha  = (const float*)d_in[19];
  const float* g_ffn  = (const float*)d_in[20];
  const float* b_ffn  = (const float*)d_in[21];

  char* ws = (char*)d_ws;
  size_t off = 0;
  auto carve = [&](size_t bytes) -> void* {
    void* p = ws + off;
    off += (bytes + 255) & ~(size_t)255;
    return p;
  };
  uint16_t* wqsB = (uint16_t*)carve((size_t)H * H * 2);
  uint16_t* wksB = (uint16_t*)carve((size_t)H * H * 2);
  uint16_t* wvsB = (uint16_t*)carve((size_t)H * H * 2);
  uint16_t* wosB = (uint16_t*)carve((size_t)H * H * 2);
  uint16_t* wqcB = (uint16_t*)carve((size_t)H * H * 2);
  uint16_t* wkcB = (uint16_t*)carve((size_t)H * H * 2);
  uint16_t* wvcB = (uint16_t*)carve((size_t)H * H * 2);
  uint16_t* wocB = (uint16_t*)carve((size_t)H * H * 2);
  uint16_t* w1B  = (uint16_t*)carve((size_t)F * H * 2);
  uint16_t* w2B  = (uint16_t*)carve((size_t)H * F * 2);
  uint16_t* encB = (uint16_t*)carve((size_t)ME * H * 2);
  uint16_t* xnB  = (uint16_t*)carve((size_t)M * H * 2);
  uint16_t* qB   = (uint16_t*)carve((size_t)M * H * 2);
  uint16_t* kB   = (uint16_t*)carve((size_t)M * H * 2);
  uint16_t* vB   = (uint16_t*)carve((size_t)M * H * 2);
  uint16_t* ctxB = (uint16_t*)carve((size_t)M * H * 2);
  float*    xres = (float*)carve((size_t)M * H * 4);
  float*    zres = (float*)carve((size_t)M * H * 4);
  uint16_t* hB   = (uint16_t*)carve((size_t)M * F * 2);

  auto cvt = [&](const float* s, uint16_t* d, int n) {
    const int n4 = n / 4;
    cvt_f32_bf16<<<(n4 + 1023) / 1024, 256, 0, stream>>>((const float4*)s, (uint2*)d, n4);
  };
  cvt(Wq_s, wqsB, H * H); cvt(Wk_s, wksB, H * H);
  cvt(Wv_s, wvsB, H * H); cvt(Wo_s, wosB, H * H);
  cvt(Wq_c, wqcB, H * H); cvt(Wk_c, wkcB, H * H);
  cvt(Wv_c, wvcB, H * H); cvt(Wo_c, wocB, H * H);
  cvt(w1, w1B, F * H);    cvt(w2, w2B, H * F);
  cvt(enc, encB, ME * H);

  // proj -> bf16 (no bias/relu/res)
  auto gemm_proj = [&](const uint16_t* A, int lda, const uint16_t* Wt,
                       uint16_t* C, int N, int K) {
    gemm_bf16_t<1, 0, 0, 0><<<dim3(N / 128, M / 128), 256, 0, stream>>>(
        A, lda, Wt, K, nullptr, C, N, nullptr, nullptr, K);
  };

  // ---- Masked self-attention (pre-norm, residual) ----
  ln_bf16<<<M / 8, 256, 0, stream>>>(input_, g_mmha, b_mmha, xnB);
  gemm_proj(xnB, H, wqsB, qB, H, H);
  gemm_proj(xnB, H, wksB, kB, H, H);
  gemm_proj(xnB, H, wvsB, vB, H, H);
  flash_attn<1><<<dim3(T / 128, Bn * NH), 256, 0, stream>>>(qB, kB, vB, ctxB, T, T);
  gemm_bf16_t<0, 0, 0, 1><<<dim3(H / 128, M / 128), 256, 0, stream>>>(   // x = input + sa
      ctxB, H, wosB, H, xres, nullptr, H, nullptr, input_, H);

  // ---- Cross attention ----
  ln_bf16<<<M / 8, 256, 0, stream>>>(xres, g_mha, b_mha, xnB);
  gemm_proj(xnB,  H, wqcB, qB, H, H);
  gemm_proj(encB, H, wkcB, kB, H, H);
  gemm_proj(encB, H, wvcB, vB, H, H);
  flash_attn<0><<<dim3(T / 128, Bn * NH), 256, 0, stream>>>(qB, kB, vB, ctxB, T, TE);
  gemm_bf16_t<0, 0, 0, 1><<<dim3(H / 128, M / 128), 256, 0, stream>>>(   // z = x + ca
      ctxB, H, wocB, H, zres, nullptr, H, nullptr, xres, H);

  // ---- FFN ----
  ln_bf16<<<M / 8, 256, 0, stream>>>(zres, g_ffn, b_ffn, xnB);
  gemm_bf16_t<1, 1, 1, 0><<<dim3(F / 128, M / 128), 256, 0, stream>>>(   // relu(fn@w1^T+b1)
      xnB, H, w1B, H, nullptr, hB, F, b1, nullptr, H);
  gemm_bf16_t<0, 0, 1, 1><<<dim3(H / 128, M / 128), 256, 0, stream>>>(   // z + h@w2^T + b2
      hB, F, w2B, F, (float*)d_out, nullptr, H, b2, zres, F);
}
